// SelfAttentionBlock_37744172597368
// MI455X (gfx1250) — compile-verified, matrix operand order checked by reading, not run on hardware
//
#include <hip/hip_runtime.h>
#include <hip/hip_bf16.h>

// ---------------------------------------------------------------------------
// CDNA5 (gfx1250) fused SAGAN self-attention, bf16 WMMA path.
// Round 2: software-pipelined WMMA K-loops, 256-wide flash chunks (half the
// barrier count, 18 WMMAs per wave per sync epoch).
// ---------------------------------------------------------------------------

typedef __attribute__((ext_vector_type(16))) __bf16        bf16x16;
typedef __attribute__((ext_vector_type(8)))  float         f32x8;
typedef __attribute__((ext_vector_type(8)))  unsigned int  u32x8;

__device__ __forceinline__ unsigned short f2bf(float f) {
    unsigned u = __builtin_bit_cast(unsigned, f);
    unsigned r = u + 0x7FFFu + ((u >> 16) & 1u);   // round-to-nearest-even
    return (unsigned short)(r >> 16);
}

// Per-lane 16-element bf16 WMMA A/B fragment: lane (idx = lane&15,
// basek = lane>=16 ? 8 : 0) holds K elements [basek, basek+8) and
// [basek+16, basek+24) of a K-contiguous 32-run -> two aligned 16B loads.
__device__ __forceinline__ bf16x16 frag_ld(const unsigned short* row, int ofs) {
    uint4 p0 = *(const uint4*)(row + ofs);
    uint4 p1 = *(const uint4*)(row + ofs + 16);
    u32x8 u = { p0.x, p0.y, p0.z, p0.w, p1.x, p1.y, p1.z, p1.w };
    return __builtin_bit_cast(bf16x16, u);
}

__device__ __forceinline__ f32x8 wmma_bf16(bf16x16 a, bf16x16 b, f32x8 c) {
    return __builtin_amdgcn_wmma_f32_16x16x32_bf16(
        false, a, false, b, (short)0, c, false, false);
}

// ---------------------------------------------------------------------------
// Kernel 1: spectral sigma.  sigma = || W @ normalize(U @ W)^T ||.
// ---------------------------------------------------------------------------
__global__ __launch_bounds__(256) void sigma_kernel(
    const float* __restrict__ f_w, const float* __restrict__ f_u,
    const float* __restrict__ g_w, const float* __restrict__ g_u,
    const float* __restrict__ h_w, const float* __restrict__ h_u,
    float* __restrict__ inv_out)
{
    __shared__ float v[256];
    __shared__ float red[256];
    const int k = blockIdx.x;
    const float* W = (k == 0) ? f_w : (k == 1) ? g_w : h_w;
    const float* U = (k == 0) ? f_u : (k == 1) ? g_u : h_u;
    const int O = (k == 2) ? 256 : 32;
    const int C = 256;
    const int t = threadIdx.x;

    float vt = 0.f;
    for (int i = 0; i < O; ++i) vt += U[i] * W[i * C + t];
    red[t] = vt * vt;
    __syncthreads();
    for (int s = 128; s > 0; s >>= 1) { if (t < s) red[t] += red[t + s]; __syncthreads(); }
    float denom = fmaxf(sqrtf(red[0]), 1e-12f);
    __syncthreads();
    v[t] = vt / denom;
    __syncthreads();

    float tv = 0.f;
    if (t < O) { for (int j = 0; j < C; ++j) tv += W[t * C + j] * v[j]; }
    red[t] = tv * tv;
    __syncthreads();
    for (int s = 128; s > 0; s >>= 1) { if (t < s) red[t] += red[t + s]; __syncthreads(); }
    if (t == 0) inv_out[k] = 1.0f / sqrtf(red[0]);
}

// ---------------------------------------------------------------------------
// Kernel 2: scaled-weight -> bf16.  Wcat rows: [0,32)=f, [32,64)=g, [64,320)=h.
// ---------------------------------------------------------------------------
__global__ __launch_bounds__(256) void wconv_kernel(
    const float* __restrict__ f_w, const float* __restrict__ g_w,
    const float* __restrict__ h_w, const float* __restrict__ inv,
    unsigned short* __restrict__ Wb)
{
    int e = (blockIdx.x * 256 + threadIdx.x) * 4;
#pragma unroll
    for (int i = 0; i < 4; ++i) {
        int ee = e + i;
        int o = ee >> 8, c = ee & 255;
        float wv, s;
        if (o < 32)      { wv = f_w[o * 256 + c];        s = inv[0]; }
        else if (o < 64) { wv = g_w[(o - 32) * 256 + c]; s = inv[1]; }
        else             { wv = h_w[(o - 64) * 256 + c]; s = inv[2]; }
        Wb[ee] = f2bf(wv * s);
    }
}

// ---------------------------------------------------------------------------
// Kernel 3: projections (x^T @ [f;g;h]^T + bias).  One 16-row n-tile of bf16
// x^T staged in LDS; 10 waves x 2 o-tiles x 8 pipelined WMMA(K=32).
// Outputs K(n,d)/Q(m,d) row-contiguous in d; V stored (c,m) row-contiguous
// in m so every later fragment is two 16B loads.
// ---------------------------------------------------------------------------
__global__ __launch_bounds__(320) void proj_kernel(
    const float* __restrict__ x, const unsigned short* __restrict__ Wb,
    const float* __restrict__ f_b, const float* __restrict__ g_b,
    const float* __restrict__ h_b,
    unsigned short* __restrict__ Kb, unsigned short* __restrict__ Qb,
    unsigned short* __restrict__ Vb)
{
    __shared__ __attribute__((aligned(16))) unsigned short xs[16 * 256];
    const int b  = blockIdx.y;
    const int n0 = blockIdx.x * 16;
    const int t  = threadIdx.x;

    if (t < 256) {
        const float* xp = x + ((size_t)b * 256 + t) * 4096 + n0;
#pragma unroll
        for (int rr = 0; rr < 4; ++rr) {
            float4 v = ((const float4*)xp)[rr];
            xs[(rr * 4 + 0) * 256 + t] = f2bf(v.x);
            xs[(rr * 4 + 1) * 256 + t] = f2bf(v.y);
            xs[(rr * 4 + 2) * 256 + t] = f2bf(v.z);
            xs[(rr * 4 + 3) * 256 + t] = f2bf(v.w);
        }
    }
    __syncthreads();

    const int w     = t >> 5;
    const int lane  = t & 31;
    const int lr    = lane & 15;
    const int basek = (lane >= 16) ? 8 : 0;
    const int hi    = (lane >= 16) ? 8 : 0;

    for (int hh = 0; hh < 2; ++hh) {
        const int o0  = (w + hh * 10) * 16;
        const int col = o0 + lr;
        const unsigned short* arow = &xs[lr * 256];
        const unsigned short* brow = Wb + (size_t)col * 256;
        f32x8 acc = {};
        // software-pipelined K loop: fetch k+1 fragments before WMMA k
        bf16x16 a = frag_ld(arow, basek);
        bf16x16 bb = frag_ld(brow, basek);
#pragma unroll
        for (int kk = 0; kk < 8; ++kk) {
            bf16x16 ac = a, bc = bb;
            if (kk < 7) {
                a  = frag_ld(arow, (kk + 1) * 32 + basek);
                bb = frag_ld(brow, (kk + 1) * 32 + basek);
            }
            acc = wmma_bf16(ac, bc, acc);
        }
        float bias = (col < 32) ? f_b[col] : (col < 64) ? g_b[col - 32] : h_b[col - 64];
#pragma unroll
        for (int r = 0; r < 8; ++r) {
            int n = n0 + r + hi;
            unsigned short v = f2bf(acc[r] + bias);
            if (col < 32)      Kb[((size_t)b * 4096 + n) * 32 + col] = v;
            else if (col < 64) Qb[((size_t)b * 4096 + n) * 32 + (col - 32)] = v;
            else               Vb[((size_t)b * 256 + (col - 64)) * 4096 + n] = v;
        }
    }
}

// ---------------------------------------------------------------------------
// Kernel 4: flash attention + fused epilogue.  Block = 8 waves = 16 query
// rows; the 16x32 K fragment (d=32 == one WMMA K step) is loaded once.
// m is processed in 256-wide chunks: each wave 2 S-WMMAs (cols w*32..+32),
// cooperative online softmax in LDS, then 8x2 pipelined PV WMMAs into its
// 32 fp32 accumulator channels.  Epilogue fuses 1/rowsum, the reference's
// raw (b,hw,c)->(b,c,h,w) view aliasing (flat = n*256+c), gamma*out + x.
// ---------------------------------------------------------------------------
__global__ __launch_bounds__(256) void attn_kernel(
    const unsigned short* __restrict__ Kb, const unsigned short* __restrict__ Qb,
    const unsigned short* __restrict__ Vb, const float* __restrict__ x,
    const float* __restrict__ gammaPtr, float* __restrict__ out)
{
    __shared__ __attribute__((aligned(16))) float          Ss[16 * 256];
    __shared__ __attribute__((aligned(16))) unsigned short Pb[16 * 256];
    __shared__ float part[16 * 16];
    __shared__ float rowmax[16], rowsum[16], rowscale[16];

    const int b  = blockIdx.y;
    const int n0 = blockIdx.x * 16;
    const int t  = threadIdx.x;
    const int w     = t >> 5;
    const int lane  = t & 31;
    const int lr    = lane & 15;
    const int basek = (lane >= 16) ? 8 : 0;
    const int hi    = (lane >= 16) ? 8 : 0;
    const int rrow  = t >> 4;            // softmax phase: row 0..15
    const int cg    = (t & 15) * 16;     // softmax phase: 16-col group

    if (t < 16) { rowmax[t] = -1e30f; rowsum[t] = 0.f; }

    // A-fragment: this block's 16 K rows (reused for every m chunk).
    bf16x16 afrag = frag_ld(Kb + ((size_t)b * 4096 + n0 + lr) * 32, basek);
    f32x8 acc0 = {}, acc1 = {};
    const int c0 = w * 32;               // wave's 32 output channels
    __syncthreads();

    for (int m0 = 0; m0 < 4096; m0 += 256) {
        // ---- S subtiles: wave w owns cols [m0 + w*32, +32) ----------------
        {
            const int mc = m0 + w * 32;
            bf16x16 bq0 = frag_ld(Qb + ((size_t)b * 4096 + mc + lr) * 32, basek);
            bf16x16 bq1 = frag_ld(Qb + ((size_t)b * 4096 + mc + 16 + lr) * 32, basek);
            f32x8 z0 = {}, z1 = {};
            f32x8 s0 = wmma_bf16(afrag, bq0, z0);
            f32x8 s1 = wmma_bf16(afrag, bq1, z1);
#pragma unroll
            for (int r = 0; r < 8; ++r) {
                Ss[(r + hi) * 256 + w * 32 + lr]      = s0[r];
                Ss[(r + hi) * 256 + w * 32 + 16 + lr] = s1[r];
            }
        }
        __syncthreads();

        // ---- running row max ---------------------------------------------
        {
            float m = -1e30f;
#pragma unroll
            for (int i = 0; i < 16; ++i) m = fmaxf(m, Ss[rrow * 256 + cg + i]);
            part[rrow * 16 + (t & 15)] = m;
        }
        __syncthreads();
        if (t < 16) {
            float cm = -1e30f;
#pragma unroll
            for (int i = 0; i < 16; ++i) cm = fmaxf(cm, part[t * 16 + i]);
            float om = rowmax[t];
            float nm = fmaxf(om, cm);
            float sc = __expf(om - nm);
            rowmax[t] = nm; rowscale[t] = sc; rowsum[t] *= sc;
        }
        __syncthreads();

        // ---- exp, P (bf16 in LDS), partial sums --------------------------
        {
            float nm = rowmax[rrow];
            float ls = 0.f;
#pragma unroll
            for (int i = 0; i < 16; ++i) {
                float e = __expf(Ss[rrow * 256 + cg + i] - nm);
                Pb[rrow * 256 + cg + i] = f2bf(e);
                ls += e;
            }
            part[rrow * 16 + (t & 15)] = ls;
        }
        __syncthreads();
        if (t < 16) {
            float s = 0.f;
#pragma unroll
            for (int i = 0; i < 16; ++i) s += part[t * 16 + i];
            rowsum[t] += s;
        }

        // ---- rescale accumulators ----------------------------------------
#pragma unroll
        for (int r = 0; r < 8; ++r) {
            float f = rowscale[r + hi];
            acc0[r] *= f;
            acc1[r] *= f;
        }

        // ---- P @ V: 8 pipelined K-chunks x 2 channel tiles ---------------
        const unsigned short* prow  = &Pb[lr * 256];
        const unsigned short* vrow0 = Vb + ((size_t)b * 256 + c0 + lr) * 4096 + m0;
        const unsigned short* vrow1 = vrow0 + (size_t)16 * 4096;
        if (m0 + 256 < 4096) {                       // prefetch next V chunk
            __builtin_prefetch(vrow0 + 256, 0, 3);
            __builtin_prefetch(vrow1 + 256, 0, 3);
        }
        bf16x16 p  = frag_ld(prow,  basek);
        bf16x16 v0 = frag_ld(vrow0, basek);
        bf16x16 v1 = frag_ld(vrow1, basek);
#pragma unroll
        for (int mm = 0; mm < 8; ++mm) {
            bf16x16 pc = p, v0c = v0, v1c = v1;
            if (mm < 7) {
                int o = (mm + 1) * 32 + basek;
                p  = frag_ld(prow,  o);
                v0 = frag_ld(vrow0, o);
                v1 = frag_ld(vrow1, o);
            }
            acc0 = wmma_bf16(pc, v0c, acc0);
            acc1 = wmma_bf16(pc, v1c, acc1);
        }
        __syncthreads();   // protect Ss/Pb/rowscale for next chunk
    }

    // ---- epilogue: /rowsum, view-aliasing residual ------------------------
    const float gamma = gammaPtr[0];
#pragma unroll
    for (int r = 0; r < 8; ++r) {
        float invs = 1.0f / rowsum[r + hi];
        int n = n0 + r + hi;
        size_t base = ((size_t)b * 4096 + n) * 256;
        size_t i0 = base + c0 + lr;
        size_t i1 = base + c0 + 16 + lr;
        out[i0] = gamma * (acc0[r] * invs) + x[i0];
        out[i1] = gamma * (acc1[r] * invs) + x[i1];
    }
}

// ---------------------------------------------------------------------------
// Launch
// ---------------------------------------------------------------------------
extern "C" void kernel_launch(void* const* d_in, const int* in_sizes, int n_in,
                              void* d_out, int out_size, void* d_ws, size_t ws_size,
                              hipStream_t stream) {
    (void)in_sizes; (void)n_in; (void)out_size; (void)ws_size;
    const float* x     = (const float*)d_in[0];
    const float* f_w   = (const float*)d_in[1];
    const float* f_b   = (const float*)d_in[2];
    const float* f_u   = (const float*)d_in[3];
    const float* g_w   = (const float*)d_in[4];
    const float* g_b   = (const float*)d_in[5];
    const float* g_u   = (const float*)d_in[6];
    const float* h_w   = (const float*)d_in[7];
    const float* h_b   = (const float*)d_in[8];
    const float* h_u   = (const float*)d_in[9];
    const float* gamma = (const float*)d_in[10];
    float* out = (float*)d_out;

    char* ws = (char*)d_ws;
    // workspace layout (bytes)
    float*          inv = (float*)(ws + 0);                       // 3 floats
    unsigned short* Wb  = (unsigned short*)(ws + 256);            // 320*256 bf16 = 163840
    unsigned short* Kb  = (unsigned short*)(ws + 164096);         // 4*4096*32  bf16 = 1 MiB
    unsigned short* Qb  = (unsigned short*)(ws + 164096 + 1048576);
    unsigned short* Vb  = (unsigned short*)(ws + 164096 + 2 * 1048576);  // 4*256*4096 bf16 = 8 MiB
    // total ~10.2 MiB

    sigma_kernel<<<3, 256, 0, stream>>>(f_w, f_u, g_w, g_u, h_w, h_u, inv);
    wconv_kernel<<<80, 256, 0, stream>>>(f_w, g_w, h_w, inv, Wb);
    proj_kernel<<<dim3(256, 4), 320, 0, stream>>>(x, Wb, f_b, g_b, h_b, Kb, Qb, Vb);
    attn_kernel<<<dim3(256, 4), 256, 0, stream>>>(Kb, Qb, Vb, x, gamma, out);
}
